// DirectedGINeWithAttention_3530463117324
// MI455X (gfx1250) — matrix operand order repaired and checked by reading, not compiled, not run on hardware
//
#include <hip/hip_runtime.h>
#include <hip/hip_bf16.h>

typedef __attribute__((ext_vector_type(16))) _Float16 v16h;
typedef __attribute__((ext_vector_type(8)))  float    v8f;

#define FNEG_MAX (-3.402823466e+38f)

// ---------------------------------------------------------------------------
// Generic WMMA GEMM:  Y[M, 16*NF] = (X [+ X2]) @ W[32*KT, WLD] (+bias)
// X row-major (stride 32*KT). Compile-time LDY/WLD/NSTORE/PREADD so the
// epilogue is a straight clause of stores with immediate offsets and the
// optional elementwise pre-add is branch-free.
// One wave = one 16-row tile; B fragments cached in VGPRs, grid-stride tiles.
// ---------------------------------------------------------------------------
template <int KT, int NF, int LDY, int WLD, int NSTORE, bool PREADD>
__global__ void gemm_wmma_kernel(const float* __restrict__ X,
                                 const float* __restrict__ X2, int M,
                                 const float* __restrict__ W,
                                 const float* __restrict__ bias,
                                 float* __restrict__ Y) {
  const int lane = threadIdx.x & 31;
  const int wave = blockIdx.x * (blockDim.x >> 5) + (threadIdx.x >> 5);
  const int nwav = gridDim.x * (blockDim.x >> 5);
  const int n  = lane & 15;   // B/C column
  const int hh = lane >> 4;   // B K-half / C row-half
  const int m  = lane & 15;   // A row
  const int g  = lane >> 4;   // A K-group
  constexpr int K = 32 * KT;

  union BF { v16h v; _Float16 h[16]; };
  BF b[KT][NF];
#pragma unroll
  for (int kb = 0; kb < KT; ++kb)
#pragma unroll
    for (int nf = 0; nf < NF; ++nf)
#pragma unroll
      for (int j = 0; j < 16; ++j) {
        int k = 32 * kb + 8 * hh + (j & 7) + 16 * (j >> 3);   // B matches A: lanes0-15 K=0..7,16..23, lanes16-31 K=8..15,24..31
        b[kb][nf].h[j] = (_Float16)W[k * WLD + 16 * nf + n];
      }
  float bcol[NF];
#pragma unroll
  for (int nf = 0; nf < NF; ++nf) bcol[nf] = bias ? bias[16 * nf + n] : 0.0f;

  const int ntiles = (M + 15) >> 4;
  for (int t = wave; t < ntiles; t += nwav) {
    const int row0 = t << 4;
    int rm = row0 + m; if (rm >= M) rm = M - 1;   // clamp (EXEC stays full for WMMA)
    const float* xr  = X + (long)rm * K;
    const float* xr2 = PREADD ? X2 + (long)rm * K : nullptr;
    union AF { v16h v; _Float16 h[16]; } a[KT];
#pragma unroll
    for (int kb = 0; kb < KT; ++kb) {
      const int c0 = 32 * kb + 8 * g;            // 16-bit A 16x32 layout
#pragma unroll
      for (int j = 0; j < 8; ++j) {
        float v = xr[c0 + j];
        if (PREADD) v += xr2[c0 + j];
        a[kb].h[j] = (_Float16)v;
      }
#pragma unroll
      for (int j = 0; j < 8; ++j) {
        float v = xr[c0 + 16 + j];
        if (PREADD) v += xr2[c0 + 16 + j];
        a[kb].h[8 + j] = (_Float16)v;
      }
    }
    const bool full = (row0 + 16 <= M);
#pragma unroll
    for (int nf = 0; nf < NF; ++nf) {
      v8f c = {};
#pragma unroll
      for (int kb = 0; kb < KT; ++kb)
        c = __builtin_amdgcn_wmma_f32_16x16x32_f16(false, a[kb].v, false, b[kb][nf].v,
                                                   (short)0, c, false, false);
      const int col = 16 * nf + n;
      if (NSTORE >= 16 * NF || col < NSTORE) {   // folds away when full-width
        const float bb = bcol[nf];
        float* yp = Y + (long)(row0 + 8 * hh) * LDY + col;
        if (full) {
#pragma unroll
          for (int r = 0; r < 8; ++r) yp[r * LDY] = c[r] + bb;   // immediate offsets
        } else {
#pragma unroll
          for (int r = 0; r < 8; ++r)
            if (row0 + r + 8 * hh < M) yp[r * LDY] = c[r] + bb;
        }
      }
    }
  }
}

// ---------------------------------------------------------------------------
// Small helpers
// ---------------------------------------------------------------------------
__global__ void fill_kernel(float* p, float v, long nelem) {
  long i = blockIdx.x * (long)blockDim.x + threadIdx.x;
  if (i < nelem) p[i] = v;
}

// copy src[srows,scols] into dst[., dld] at column offset coff
__global__ void pack_cols_kernel(const float* src, int srows, int scols,
                                 float* dst, int dld, int coff) {
  int i = blockIdx.x * blockDim.x + threadIdx.x;
  if (i >= srows * scols) return;
  int r = i / scols, c = i % scols;
  dst[r * dld + coff + c] = src[i];
}

__device__ inline void atomicMaxF(float* addr, float val) {
  int* ai = (int*)addr;
  int old = __float_as_int(*addr);
  while (__int_as_float(old) < val) {
    int assumed = old;
    old = atomicCAS(ai, assumed, __float_as_int(val));
    if (old == assumed) break;
  }
}

// ---------------------------------------------------------------------------
// Attention: logits (+leaky relu) and running segment max (grouped by dst)
// nproj[N,16] = [xWsrc_in | xWdst_in | xWsrc_out | xWdst_out]
// eproj[E,8]  = [eWedge_in | eWedge_out]
// ---------------------------------------------------------------------------
__global__ void att_logits_max_kernel(const int* __restrict__ src, const int* __restrict__ dst,
                                      const float* __restrict__ nproj, const float* __restrict__ eproj,
                                      const float* __restrict__ bias_in, const float* __restrict__ bias_out,
                                      float* __restrict__ alog, float* __restrict__ smax, long E) {
  long e = blockIdx.x * (long)blockDim.x + threadIdx.x;
  if (e >= E) return;
  int s = src[e], d = dst[e];
  const float* ps = nproj + (long)s * 16;
  const float* pd = nproj + (long)d * 16;
  const float* ep = eproj + e * 8;
#pragma unroll
  for (int h = 0; h < 4; ++h) {
    float ain = ps[h] + pd[4 + h] + ep[h] + bias_in[h];
    ain = ain > 0.0f ? ain : 0.2f * ain;                       // leaky_relu(0.2)
    float aout = pd[8 + h] + ps[12 + h] + ep[4 + h] + bias_out[h];
    aout = aout > 0.0f ? aout : 0.2f * aout;
    alog[e * 8 + h] = ain;
    alog[e * 8 + 4 + h] = aout;
    atomicMaxF(&smax[(long)d * 8 + h], ain);                   // both grouped by dst
    atomicMaxF(&smax[(long)d * 8 + 4 + h], aout);
  }
}

__global__ void att_exp_sum_kernel(const int* __restrict__ dst, float* __restrict__ alog,
                                   const float* __restrict__ smax, float* __restrict__ ssum, long E) {
  long e = blockIdx.x * (long)blockDim.x + threadIdx.x;
  if (e >= E) return;
  int d = dst[e];
#pragma unroll
  for (int h = 0; h < 8; ++h) {
    float v = expf(alog[e * 8 + h] - smax[(long)d * 8 + h]);
    alog[e * 8 + h] = v;
    atomicAdd(&ssum[(long)d * 8 + h], v);
  }
}

__global__ void att_coef_kernel(const int* __restrict__ dst, const float* __restrict__ alog,
                                const float* __restrict__ ssum, float* __restrict__ coef, long E) {
  long e = blockIdx.x * (long)blockDim.x + threadIdx.x;
  if (e >= E) return;
  int d = dst[e];
  float ci = 0.0f, co = 0.0f;
#pragma unroll
  for (int h = 0; h < 4; ++h) {
    ci += alog[e * 8 + h]     / (ssum[(long)d * 8 + h]     + 1e-16f);
    co += alog[e * 8 + 4 + h] / (ssum[(long)d * 8 + 4 + h] + 1e-16f);
  }
  coef[2 * e]     = ci * 0.25f;   // mean over heads
  coef[2 * e + 1] = co * 0.25f;
}

// Fused GINE message + scatter-add (lane-per-feature, coalesced atomics).
// in : msg = relu(x[s] + ea*ci)  -> agg_in[d]
// out: msg = relu(x[d] + ea*co)  -> agg_out[s]
__global__ void gine_agg_kernel(const int* __restrict__ src, const int* __restrict__ dst,
                                const float* __restrict__ xe, const float* __restrict__ ea,
                                const float* __restrict__ coef,
                                float* __restrict__ agg_in, float* __restrict__ agg_out, long E) {
  long idx = blockIdx.x * (long)blockDim.x + threadIdx.x;
  long e = idx >> 5;
  int  k = (int)(idx & 31);
  if (e >= E) return;
  int s = src[e], d = dst[e];
  float ev = ea[e * 32 + k];
  float ci = coef[2 * e], co = coef[2 * e + 1];
  float mi = fmaxf(xe[(long)s * 32 + k] + ev * ci, 0.0f);
  float mo = fmaxf(xe[(long)d * 32 + k] + ev * co, 0.0f);
  atomicAdd(&agg_in[(long)d * 32 + k], mi);
  atomicAdd(&agg_out[(long)s * 32 + k], mo);
}

// ---------------------------------------------------------------------------
// BatchNorm (training-mode batch stats): column sums/sumsq then apply.
// X is [M,C] contiguous (ld==C). stats[0..C)=sum, stats[C..2C)=sumsq.
// ---------------------------------------------------------------------------
__global__ void col_stats_kernel(const float* __restrict__ X, int M, int C, float* stats) {
  __shared__ float sh[512];
  int c = threadIdx.x % C;
  int rg = threadIdx.x / C;
  int nrg = blockDim.x / C;
  float s = 0.0f, sq = 0.0f;
  for (long r = (long)blockIdx.x * nrg + rg; r < M; r += (long)gridDim.x * nrg) {
    float v = X[r * C + c];
    s += v; sq += v * v;
  }
  sh[threadIdx.x] = s;
  sh[256 + threadIdx.x] = sq;
  __syncthreads();
  if (rg == 0) {
    for (int i = 1; i < nrg; ++i) { s += sh[i * C + c]; sq += sh[256 + i * C + c]; }
    atomicAdd(&stats[c], s);
    atomicAdd(&stats[C + c], sq);
  }
}

__global__ void bn_relu_kernel(float* __restrict__ X, int M, int C,
                               const float* __restrict__ stats,
                               const float* __restrict__ g, const float* __restrict__ b) {
  long i = blockIdx.x * (long)blockDim.x + threadIdx.x;
  if (i >= (long)M * C) return;
  int c = (int)(i % C);
  float mu = stats[c] / (float)M;
  float var = stats[C + c] / (float)M - mu * mu;       // biased variance
  float v = (X[i] - mu) * rsqrtf(var + 1e-5f) * g[c] + b[c];
  X[i] = fmaxf(v, 0.0f);
}

// tsum = relu(bnA(xin)) + relu(bnB(xout)), both C=32
__global__ void combine_bn2_kernel(const float* __restrict__ xin, const float* __restrict__ xout,
                                   const float* __restrict__ statsA, const float* __restrict__ gA,
                                   const float* __restrict__ bA,
                                   const float* __restrict__ statsB, const float* __restrict__ gB,
                                   const float* __restrict__ bB,
                                   float* __restrict__ tsum, int M) {
  long i = blockIdx.x * (long)blockDim.x + threadIdx.x;
  if (i >= (long)M * 32) return;
  int c = (int)(i & 31);
  float muA = statsA[c] / (float)M;
  float vaA = statsA[32 + c] / (float)M - muA * muA;
  float vA = fmaxf((xin[i] - muA) * rsqrtf(vaA + 1e-5f) * gA[c] + bA[c], 0.0f);
  float muB = statsB[c] / (float)M;
  float vaB = statsB[32 + c] / (float)M - muB * muB;
  float vB = fmaxf((xout[i] - muB) * rsqrtf(vaB + 1e-5f) * gB[c] + bB[c], 0.0f);
  tsum[i] = vA + vB;
}

// xn = (xe + relu(bn(tsum))) * 0.5
__global__ void residual_bn_kernel(const float* __restrict__ xe, const float* __restrict__ tsum,
                                   const float* __restrict__ stats,
                                   const float* __restrict__ g, const float* __restrict__ b,
                                   float* __restrict__ xn, int M) {
  long i = blockIdx.x * (long)blockDim.x + threadIdx.x;
  if (i >= (long)M * 32) return;
  int c = (int)(i & 31);
  float mu = stats[c] / (float)M;
  float var = stats[32 + c] / (float)M - mu * mu;
  float v = (tsum[i] - mu) * rsqrtf(var + 1e-5f) * g[c] + b[c];
  v = fmaxf(v, 0.0f);
  xn[i] = (xe[i] + v) * 0.5f;
}

// ---------------------------------------------------------------------------
// Host side
// ---------------------------------------------------------------------------
static inline int cdiv_l(long a, int b) { return (int)((a + b - 1) / b); }

template <int KT, int NF, int LDY, int WLD, int NSTORE, bool PREADD>
static void launch_gemm(const float* X, const float* X2, int M, const float* W,
                        const float* bias, float* Y, hipStream_t s) {
  int ntiles = (M + 15) / 16;
  int blocks = (ntiles + 7) / 8;           // 8 waves / block (256 threads, wave32)
  if (blocks > 16384) blocks = 16384;
  gemm_wmma_kernel<KT, NF, LDY, WLD, NSTORE, PREADD><<<blocks, 256, 0, s>>>(X, X2, M, W, bias, Y);
}

extern "C" void kernel_launch(void* const* d_in, const int* in_sizes, int n_in,
                              void* d_out, int out_size, void* d_ws, size_t ws_size,
                              hipStream_t stream) {
  (void)n_in; (void)ws_size; (void)out_size;
  const int  N = in_sizes[0] / 32;
  const long E = in_sizes[1] / 2;

  const float* x_in   = (const float*)d_in[0];
  const int*   eidx   = (const int*)d_in[1];
  const float* ea_in  = (const float*)d_in[2];
  const int* srcI = eidx;           // edge_index[0,:]
  const int* dstI = eidx + E;       // edge_index[1,:]

  const float* eembW = (const float*)d_in[3];
  const float* eembB = (const float*)d_in[4];
  const float* mlpW  = (const float*)d_in[83];
  const float* mlpB  = (const float*)d_in[84];
  const float* nembW = (const float*)d_in[85];
  const float* nembB = (const float*)d_in[86];

  // workspace carve-up (floats)
  float* ws = (float*)d_ws;
  size_t o = 0;
  auto A = [&](size_t n) { float* p = ws + o; o += n; return p; };
  float* xeA  = A((size_t)N * 32);
  float* xeB  = A((size_t)N * 32);
  float* ea   = A((size_t)E * 32);
  float* eprj = A((size_t)E * 8);
  float* alog = A((size_t)E * 8);
  float* coef = A((size_t)E * 2);
  float* nprj = A((size_t)N * 16);
  float* smax = A((size_t)N * 8);
  float* ssum = A((size_t)N * 8);
  float* aggI = A((size_t)N * 32);
  float* aggO = A((size_t)N * 32);
  float* tbuf = A((size_t)N * 32);
  float* h64  = A((size_t)N * 64);
  float* xin  = A((size_t)N * 32);
  float* xout = A((size_t)N * 32);
  float* statsA = A(128);
  float* statsB = A(128);
  float* wcat  = A(512);    // [32,16] = [Wsrc_in|Wdst_in|Wsrc_out|Wdst_out]
  float* wecat = A(512);    // [32,16] = [Wedge_in|Wedge_out|0]
  float* wmlp  = A(512);    // [32,16] padded
  float* bmlp  = A(16);

  // embeddings (WMMA)
  launch_gemm<1, 2, 32, 32, 32, false>(x_in, nullptr, N, nembW, nembB, xeA, stream);
  launch_gemm<1, 2, 32, 32, 32, false>(ea_in, nullptr, (int)E, eembW, eembB, ea, stream);

  float* xe = xeA;
  float* xn = xeB;

  const int TB = 256;
  const int gE   = cdiv_l(E, TB);
  const int gE32 = cdiv_l(E * 32, TB);
  const int gN8  = cdiv_l((long)N * 8, TB);
  const int gN32 = cdiv_l((long)N * 32, TB);
  const int gN64 = cdiv_l((long)N * 64, TB);

  for (int l = 0; l < 3; ++l) {
    const int L = 5 + 26 * l;
    const float* WdstI_  = (const float*)d_in[L + 0];
    const float* WedgeI_ = (const float*)d_in[L + 1];
    const float* WsrcI_  = (const float*)d_in[L + 2];
    const float* biasI_  = (const float*)d_in[L + 3];
    const float* WdstO_  = (const float*)d_in[L + 4];
    const float* WedgeO_ = (const float*)d_in[L + 5];
    const float* WsrcO_  = (const float*)d_in[L + 6];
    const float* biasO_  = (const float*)d_in[L + 7];
    const float* bn_b = (const float*)d_in[L + 8];
    const float* bn_g = (const float*)d_in[L + 9];
    // conv_in / conv_out leaves (sorted: be1, be2, g1, g2, lin1W, lin1b, lin2W, lin2b)
    const float* be1I = (const float*)d_in[L + 10];
    const float* be2I = (const float*)d_in[L + 11];
    const float* g1I  = (const float*)d_in[L + 12];
    const float* g2I  = (const float*)d_in[L + 13];
    const float* l1WI = (const float*)d_in[L + 14];
    const float* l1bI = (const float*)d_in[L + 15];
    const float* l2WI = (const float*)d_in[L + 16];
    const float* l2bI = (const float*)d_in[L + 17];
    const float* be1O = (const float*)d_in[L + 18];
    const float* be2O = (const float*)d_in[L + 19];
    const float* g1O  = (const float*)d_in[L + 20];
    const float* g2O  = (const float*)d_in[L + 21];
    const float* l1WO = (const float*)d_in[L + 22];
    const float* l1bO = (const float*)d_in[L + 23];
    const float* l2WO = (const float*)d_in[L + 24];
    const float* l2bO = (const float*)d_in[L + 25];

    // pack attention weight blocks
    pack_cols_kernel<<<1, 128, 0, stream>>>(WsrcI_, 32, 4, wcat, 16, 0);
    pack_cols_kernel<<<1, 128, 0, stream>>>(WdstI_, 32, 4, wcat, 16, 4);
    pack_cols_kernel<<<1, 128, 0, stream>>>(WsrcO_, 32, 4, wcat, 16, 8);
    pack_cols_kernel<<<1, 128, 0, stream>>>(WdstO_, 32, 4, wcat, 16, 12);
    hipMemsetAsync(wecat, 0, 512 * sizeof(float), stream);
    pack_cols_kernel<<<1, 128, 0, stream>>>(WedgeI_, 32, 4, wecat, 16, 0);
    pack_cols_kernel<<<1, 128, 0, stream>>>(WedgeO_, 32, 4, wecat, 16, 4);

    // projections (WMMA)
    launch_gemm<1, 1, 16, 16, 16, false>(xe, nullptr, N, wcat, nullptr, nprj, stream);
    launch_gemm<1, 1, 8, 16, 8, false>(ea, nullptr, (int)E, wecat, nullptr, eprj, stream);

    // segment softmax grouped by dst (both directions)
    fill_kernel<<<gN8, TB, 0, stream>>>(smax, FNEG_MAX, (long)N * 8);
    hipMemsetAsync(ssum, 0, (size_t)N * 8 * sizeof(float), stream);
    att_logits_max_kernel<<<gE, TB, 0, stream>>>(srcI, dstI, nprj, eprj, biasI_, biasO_,
                                                 alog, smax, E);
    att_exp_sum_kernel<<<gE, TB, 0, stream>>>(dstI, alog, smax, ssum, E);
    att_coef_kernel<<<gE, TB, 0, stream>>>(dstI, alog, ssum, coef, E);

    // fused message + scatter aggregation for both directions
    hipMemsetAsync(aggI, 0, (size_t)N * 32 * sizeof(float), stream);
    hipMemsetAsync(aggO, 0, (size_t)N * 32 * sizeof(float), stream);
    gine_agg_kernel<<<gE32, TB, 0, stream>>>(srcI, dstI, xe, ea, coef, aggI, aggO, E);

    // ---- conv_in: mlp(x + agg_in), pre-add fused into lin1 A-load ----
    launch_gemm<1, 4, 64, 64, 64, true>(xe, aggI, N, l1WI, l1bI, h64, stream);
    hipMemsetAsync(statsA, 0, 128 * sizeof(float), stream);
    col_stats_kernel<<<768, 256, 0, stream>>>(h64, N, 64, statsA);
    bn_relu_kernel<<<gN64, TB, 0, stream>>>(h64, N, 64, statsA, g1I, be1I);
    launch_gemm<2, 2, 32, 32, 32, false>(h64, nullptr, N, l2WI, l2bI, xin, stream);
    hipMemsetAsync(statsA, 0, 128 * sizeof(float), stream);
    col_stats_kernel<<<768, 256, 0, stream>>>(xin, N, 32, statsA);   // xin left raw

    // ---- conv_out: mlp(x + agg_out) ----
    launch_gemm<1, 4, 64, 64, 64, true>(xe, aggO, N, l1WO, l1bO, h64, stream);
    hipMemsetAsync(statsB, 0, 128 * sizeof(float), stream);
    col_stats_kernel<<<768, 256, 0, stream>>>(h64, N, 64, statsB);
    bn_relu_kernel<<<gN64, TB, 0, stream>>>(h64, N, 64, statsB, g1O, be1O);
    launch_gemm<2, 2, 32, 32, 32, false>(h64, nullptr, N, l2WO, l2bO, xout, stream);
    hipMemsetAsync(statsB, 0, 128 * sizeof(float), stream);
    col_stats_kernel<<<768, 256, 0, stream>>>(xout, N, 32, statsB);  // xout left raw

    // ---- combine: tsum = relu(bn(xin)) + relu(bn(xout)) in one pass ----
    combine_bn2_kernel<<<gN32, TB, 0, stream>>>(xin, xout, statsA, g2I, be2I,
                                                statsB, g2O, be2O, tbuf, N);
    hipMemsetAsync(statsA, 0, 128 * sizeof(float), stream);
    col_stats_kernel<<<768, 256, 0, stream>>>(tbuf, N, 32, statsA);
    residual_bn_kernel<<<gN32, TB, 0, stream>>>(xe, tbuf, statsA, bn_g, bn_b, xn, N);

    float* tmp = xe; xe = xn; xn = tmp;
  }

  // output head: [N,32] @ [32,1] + b  (padded to 16 cols for WMMA, store 1)
  hipMemsetAsync(wmlp, 0, 512 * sizeof(float), stream);
  hipMemsetAsync(bmlp, 0, 16 * sizeof(float), stream);
  pack_cols_kernel<<<1, 128, 0, stream>>>(mlpW, 32, 1, wmlp, 16, 0);
  pack_cols_kernel<<<1, 128, 0, stream>>>(mlpB, 1, 1, bmlp, 16, 0);
  launch_gemm<1, 1, 1, 16, 1, false>(xe, nullptr, N, wmlp, bmlp, (float*)d_out, stream);
}
